// MultiHeadAttention_91285234909775
// MI455X (gfx1250) — matrix operand order repaired and checked by reading, not compiled, not run on hardware
//
#include <hip/hip_runtime.h>
#include <hip/hip_bf16.h>
#include <stdint.h>

// ---------------------------------------------------------------------------
// MI455X (gfx1250) multi-head attention, f16 WMMA path, wave32.
//  - projections: ping-pong (2x unrolled) register-double-buffered WMMA GEMM,
//    no inter-buffer moves, global prefetch ahead of the pipeline
//  - attention : async-to-LDS (ASYNCcnt) double-buffered K/V/enc staging
// ---------------------------------------------------------------------------

#define B_    2
#define L_    2048
#define DIM_  1024
#define H_    16
#define DH_   64
#define POS_  16
#define OUTD_ 1040           // DIM + POS
#define SCALE_INV_ 0.125f    // 1 / round(sqrt(64),2) = 1/8

typedef __attribute__((ext_vector_type(16))) _Float16 v16h;
typedef __attribute__((ext_vector_type(8)))  float    v8f;

union Frag16 {
    v16h     h;
    uint32_t u[8];
};

// ----- WMMA wrapper: D = A(16x32 f16) * B(32x16 f16) + C(16x16 f32) --------
__device__ __forceinline__ v8f wmma_f16(v16h a, v16h b, v8f c) {
    return __builtin_amdgcn_wmma_f32_16x16x32_f16(
        /*neg_a=*/false, a, /*neg_b=*/false, b,
        /*c_mod=*/(short)0, c, /*reuse_a=*/false, /*reuse_b=*/false);
}

// ----- A-fragment (16x32, MxK) from row-major storage ----------------------
// element (m,k) = base[m*ld + k].  ISA layout: lanes 0-15 -> M=lane,
// VGPR j<4: K = 2j + 8*half ; j>=4: K = 16 + 2(j-4) + 8*half.
__device__ __forceinline__ v16h load_a_rowmajor(const _Float16* base, int ld, int lane) {
    Frag16 f;
    const int m = lane & 15, half = lane >> 4;
    const _Float16* p = base + m * ld + 8 * half;
#pragma unroll
    for (int j = 0; j < 4; ++j)
        f.u[j] = *(const uint32_t*)(p + 2 * j);
#pragma unroll
    for (int j = 0; j < 4; ++j)
        f.u[4 + j] = *(const uint32_t*)(p + 16 + 2 * j);
    return f.h;
}

// ----- B-fragment (32x16, KxN) where memory is N-major ---------------------
// element (k,n) = base[n*ld + k]   (covers X*W^T with row-major W, Q*K^T with
// row-major K, P*V with d-major V, P*enc with p-major encT).
// ISA layout: lane -> N = lane&15, VGPR j<4: K = 2j + 8*half ; j>=4: K = 16 + 2(j-4) + 8*half.
__device__ __forceinline__ v16h load_b_ntrans(const _Float16* base, int ld, int lane) {
    Frag16 f;
    const int n = lane & 15, half = lane >> 4;
    const _Float16* p = base + n * ld + 8 * half;
#pragma unroll
    for (int j = 0; j < 4; ++j)
        f.u[j] = *(const uint32_t*)(p + 2 * j);
#pragma unroll
    for (int j = 0; j < 4; ++j)
        f.u[4 + j] = *(const uint32_t*)(p + 16 + 2 * j);
    return f.h;
}

// ----- CDNA5 async global->LDS copy (ASYNCcnt-tracked, bypasses VGPRs) -----
__device__ __forceinline__ void async_copy_b128(uint32_t lds_addr, uint64_t gaddr) {
    asm volatile("global_load_async_to_lds_b128 %0, %1, off"
                 :: "v"(lds_addr), "v"(gaddr)
                 : "memory");
}
__device__ __forceinline__ void wait_async_pending18() {
    asm volatile("s_wait_asynccnt 18" ::: "memory");
}
__device__ __forceinline__ void wait_async_all() {
    asm volatile("s_wait_asynccnt 0" ::: "memory");
}
__device__ __forceinline__ uint32_t lds_off(const void* p) {
    // local addresses are workgroup-relative in the low 32 bits
    return (uint32_t)(uintptr_t)p;
}

// ---------------------------------------------------------------------------
// Kernel 1: f32 -> f16 conversion / relayout.
//   qf/kf/vf : (B*L, 1024) f16   (pos dims stripped)
//   W*16     : (1024,1024) f16 row-major
//   encT     : (B, 16, L)  f16   (transposed encoding, K-contiguous for WMMA B)
//   encacc   : (B*L, 16)   f32   zeroed
// ---------------------------------------------------------------------------
__global__ void convert_kernel(const float* __restrict__ q, const float* __restrict__ k,
                               const float* __restrict__ v,
                               const float* __restrict__ Wq, const float* __restrict__ Wk,
                               const float* __restrict__ Wv, const float* __restrict__ Wo,
                               _Float16* __restrict__ qf, _Float16* __restrict__ kf,
                               _Float16* __restrict__ vf,
                               _Float16* __restrict__ Wq16, _Float16* __restrict__ Wk16,
                               _Float16* __restrict__ Wv16, _Float16* __restrict__ Wo16,
                               _Float16* __restrict__ encT, float* __restrict__ encacc) {
    const int i = blockIdx.x * blockDim.x + threadIdx.x;   // [0, B*L*DIM)
    const int row = i >> 10;       // DIM = 1024
    const int d   = i & 1023;
    const int src = row * (DIM_ + POS_) + d;
    qf[i] = (_Float16)q[src];
    kf[i] = (_Float16)k[src];
    vf[i] = (_Float16)v[src];
    if (i < DIM_ * DIM_) {
        Wq16[i] = (_Float16)Wq[i];
        Wk16[i] = (_Float16)Wk[i];
        Wv16[i] = (_Float16)Wv[i];
        Wo16[i] = (_Float16)Wo[i];
    }
    if (i < B_ * L_ * POS_) {
        const int p  = i & 15;
        const int bl = i >> 4;
        const int b  = bl >> 11;       // L = 2048
        const int l  = bl & 2047;
        encT[((size_t)b * POS_ + p) * L_ + l] =
            (_Float16)k[(size_t)(b * L_ + l) * (DIM_ + POS_) + DIM_ + p];
        encacc[i] = 0.0f;
    }
}

// ---------------------------------------------------------------------------
// Kernel 2: generic projection GEMM  Y = X * W^T + bias.
//   M = B*L = 4096, N = K = 1024.  One wave computes a 16x64 tile.
//   K-loop processes 64 (two 32-chunks) per iteration with two disjoint
//   fragment register sets (ping-pong) -> zero inter-buffer moves, loads of
//   set X overlap the WMMAs consuming set Y.
//   mode 0/1 : write f16 head-major (b,h,l,d)           (Q, K)
//   mode 2   : write f16 head-major transposed (b,h,d,l) (V)
//   mode 3   : write f32 into output, row stride 1040    (final projection)
// ---------------------------------------------------------------------------
__global__ void proj_gemm(const _Float16* __restrict__ X, const _Float16* __restrict__ W,
                          const float* __restrict__ bias,
                          _Float16* __restrict__ out16, float* __restrict__ out32,
                          int mode) {
    const int wave = (blockIdx.x * blockDim.x + threadIdx.x) >> 5;
    const int lane = threadIdx.x & 31;
    const int NT   = DIM_ / 64;                 // 16 col-tiles of 64
    const int m0   = (wave / NT) * 16;
    const int n0   = (wave % NT) * 64;

    const _Float16* Xrow = X + (size_t)m0 * DIM_;
    const _Float16* Wrow = W + (size_t)n0 * DIM_;

    v8f acc[4] = {};

    // prologue: set A <- chunk 0
    v16h aA = load_a_rowmajor(Xrow, DIM_, lane);
    v16h bA[4], bB[4];
    v16h aB = aA;
#pragma unroll
    for (int c = 0; c < 4; ++c)
        bA[c] = load_b_ntrans(Wrow + (size_t)(16 * c) * DIM_, DIM_, lane);

    for (int kk = 0; kk < DIM_; kk += 64) {
        // set B <- chunk kk+32 (always exists: kk <= 960 -> 992 < 1024)
        {
            const int kn = kk + 32;
            __builtin_prefetch(Xrow + (lane & 15) * DIM_ + kn + 64, 0, 3);
            __builtin_prefetch(Wrow + (size_t)(lane & 15) * DIM_ + kn + 64, 0, 3);
            aB = load_a_rowmajor(Xrow + kn, DIM_, lane);
#pragma unroll
            for (int c = 0; c < 4; ++c)
                bB[c] = load_b_ntrans(Wrow + (size_t)(16 * c) * DIM_ + kn, DIM_, lane);
        }
        // compute chunk kk with set A
#pragma unroll
        for (int c = 0; c < 4; ++c)
            acc[c] = wmma_f16(aA, bA[c], acc[c]);

        // set A <- chunk kk+64 (skip past the end)
        if (kk + 64 < DIM_) {
            const int kn = kk + 64;
            aA = load_a_rowmajor(Xrow + kn, DIM_, lane);
#pragma unroll
            for (int c = 0; c < 4; ++c)
                bA[c] = load_b_ntrans(Wrow + (size_t)(16 * c) * DIM_ + kn, DIM_, lane);
        }
        // compute chunk kk+32 with set B
#pragma unroll
        for (int c = 0; c < 4; ++c)
            acc[c] = wmma_f16(aB, bB[c], acc[c]);
    }

    const int half = lane >> 4, nl = lane & 15;
#pragma unroll
    for (int c = 0; c < 4; ++c) {
        const int   n  = n0 + 16 * c + nl;
        const float bv = bias[n];
#pragma unroll
        for (int r = 0; r < 8; ++r) {
            const int   m   = m0 + r + 8 * half;        // global row in [0, B*L)
            const float val = acc[c][r] + bv;
            if (mode == 3) {
                out32[(size_t)m * OUTD_ + n] = val;
            } else {
                const int b = m >> 11, l = m & 2047;    // L = 2048
                const int h = n >> 6,  d = n & 63;      // DH = 64
                if (mode == 2)
                    out16[((size_t)(b * H_ + h) * DH_ + d) * L_ + l] = (_Float16)val;
                else
                    out16[((size_t)(b * H_ + h) * L_ + l) * DH_ + d] = (_Float16)val;
            }
        }
    }
}

// ---------------------------------------------------------------------------
// Kernel 3: causal flash attention, one wave per 16-query tile.
//   K/V/enc chunks (32 keys) streamed into a 2-deep LDS ring with
//   GLOBAL_LOAD_ASYNC_TO_LDS_B128 (18 async ops/chunk), overlapped with WMMA.
//   Per 32-key chunk: 4 WMMA (QK^T) + 4 WMMA (P*V) + 1 WMMA (P*enc).
//   Causality handled analytically -> only floor(q0/32)+1 chunks visited.
// ---------------------------------------------------------------------------
__global__ void attn_kernel(const _Float16* __restrict__ Qh, const _Float16* __restrict__ Kh,
                            const _Float16* __restrict__ Vt, const _Float16* __restrict__ encT,
                            _Float16* __restrict__ ctx16, float* __restrict__ encacc) {
    __shared__ _Float16 ldsK[2][32 * DH_];    // 32 keys x 64 d      (2 x 4 KB)
    __shared__ _Float16 ldsV[2][DH_ * 32];    // 64 d    x 32 keys   (2 x 4 KB)
    __shared__ _Float16 ldsE[2][POS_ * 32];   // 16 p    x 32 keys   (2 x 1 KB)
    __shared__ _Float16 ldsP[16 * 32];        // P tile D-layout -> A-layout

    const int lane = threadIdx.x & 31;
    const int bid  = blockIdx.x;              // B*H*(L/16) = 4096
    const int qt   = bid & 127;               // L/16 = 128
    const int h    = (bid >> 7) & 15;
    const int b    = bid >> 11;
    const int q0   = qt * 16;

    const _Float16* Qb = Qh + ((size_t)(b * H_ + h) * L_) * DH_;
    const _Float16* Kb = Kh + ((size_t)(b * H_ + h) * L_) * DH_;
    const _Float16* Vb = Vt + ((size_t)(b * H_ + h) * DH_) * L_;
    const _Float16* Eb = encT + (size_t)b * POS_ * L_;

    // Q A-fragments are loop-invariant (16 rows x 64 d -> two 16x32 frags)
    const v16h qa0 = load_a_rowmajor(Qb + (size_t)q0 * DH_ + 0,  DH_, lane);
    const v16h qa1 = load_a_rowmajor(Qb + (size_t)q0 * DH_ + 32, DH_, lane);

    const int half = lane >> 4, nl = lane & 15;

    // ---- async stage of one 32-key chunk into LDS buffer `buf` ------------
    auto issue_chunk = [&](int buf, int j0) {
        // K: lane copies key row (j0+lane): 128 B = 8 x b128
        {
            const uint64_t src = (uint64_t)(uintptr_t)(Kb + (size_t)(j0 + lane) * DH_);
            const uint32_t dst = lds_off(&ldsK[buf][lane * DH_]);
#pragma unroll
            for (int t = 0; t < 8; ++t)
                async_copy_b128(dst + 16 * t, src + 16 * t);
        }
        // V: lane copies d-rows lane and lane+32: 2 x 64 B = 8 x b128
#pragma unroll
        for (int g = 0; g < 2; ++g) {
            const int d = lane + 32 * g;
            const uint64_t src = (uint64_t)(uintptr_t)(Vb + (size_t)d * L_ + j0);
            const uint32_t dst = lds_off(&ldsV[buf][d * 32]);
#pragma unroll
            for (int t = 0; t < 4; ++t)
                async_copy_b128(dst + 16 * t, src + 16 * t);
        }
        // enc: lane copies half a p-row (32 B) = 2 x b128
        {
            const int p   = lane >> 1;
            const int off = (lane & 1) * 16;            // f16 elements
            const uint64_t src = (uint64_t)(uintptr_t)(Eb + (size_t)p * L_ + j0 + off);
            const uint32_t dst = lds_off(&ldsE[buf][p * 32 + off]);
            async_copy_b128(dst,      src);
            async_copy_b128(dst + 16, src + 16);
        }
    };

    float mrun[8], lrun[8];
#pragma unroll
    for (int r = 0; r < 8; ++r) { mrun[r] = -1.0e30f; lrun[r] = 0.0f; }
    v8f O[4] = {};
    v8f Oe   = {};

    const int nChunks = (qt >> 1) + 1;          // keys 0 .. q0+15, 32 at a time
    int buf = 0;
    issue_chunk(0, 0);

    for (int jc = 0; jc < nChunks; ++jc) {
        const int j0 = jc * 32;

        // keep the next chunk in flight while we compute this one
        if (jc + 1 < nChunks) {
            issue_chunk(buf ^ 1, j0 + 32);
            wait_async_pending18();             // chunk `jc` complete (in-order)
        } else {
            wait_async_all();
        }

        // ---- S = Q * K^T for two 16-key tiles (B-frags from LDS) ----------
        v8f S0 = {}, S1 = {};
        {
            v16h kb = load_b_ntrans(&ldsK[buf][0],            DH_, lane);
            S0 = wmma_f16(qa0, kb, S0);
            kb = load_b_ntrans(&ldsK[buf][32],                DH_, lane);
            S0 = wmma_f16(qa1, kb, S0);
            kb = load_b_ntrans(&ldsK[buf][16 * DH_],          DH_, lane);
            S1 = wmma_f16(qa0, kb, S1);
            kb = load_b_ntrans(&ldsK[buf][16 * DH_ + 32],     DH_, lane);
            S1 = wmma_f16(qa1, kb, S1);
        }

        // ---- scale, causal mask, online softmax ---------------------------
        const int col0 = j0 + nl, col1 = col0 + 16;
        float p0[8], p1[8], alpha[8];
#pragma unroll
        for (int r = 0; r < 8; ++r) {
            const int row = q0 + r + 8 * half;
            float s0 = S0[r] * SCALE_INV_; if (col0 > row) s0 = -1.0e9f;
            float s1 = S1[r] * SCALE_INV_; if (col1 > row) s1 = -1.0e9f;

            float tm = fmaxf(s0, s1);
#pragma unroll
            for (int ms = 1; ms < 16; ms <<= 1)      // row spans 16 lanes of a half
                tm = fmaxf(tm, __shfl_xor(tm, ms, 32));

            const float mnew = fmaxf(mrun[r], tm);
            alpha[r] = __expf(mrun[r] - mnew);
            mrun[r]  = mnew;
            p0[r] = __expf(s0 - mnew);
            p1[r] = __expf(s1 - mnew);

            float rs = p0[r] + p1[r];
#pragma unroll
            for (int ms = 1; ms < 16; ms <<= 1)
                rs += __shfl_xor(rs, ms, 32);
            lrun[r] = lrun[r] * alpha[r] + rs;
        }

        // ---- rescale running accumulators ---------------------------------
#pragma unroll
        for (int r = 0; r < 8; ++r) {
#pragma unroll
            for (int c = 0; c < 4; ++c) O[c][r] *= alpha[r];
            Oe[r] *= alpha[r];
        }

        // ---- P: D-layout -> A-fragment via LDS (intra-wave, DScnt-ordered)
#pragma unroll
        for (int r = 0; r < 8; ++r) {
            const int m = r + 8 * half;
            ldsP[m * 32 + nl]      = (_Float16)p0[r];
            ldsP[m * 32 + 16 + nl] = (_Float16)p1[r];
        }
        const v16h pa = load_a_rowmajor(ldsP, 32, lane);

        // ---- O += P * V ; Oe += P * enc  (B-frags from LDS) ---------------
#pragma unroll
        for (int c = 0; c < 4; ++c) {
            const v16h vb = load_b_ntrans(&ldsV[buf][(16 * c) * 32], 32, lane);
            O[c] = wmma_f16(pa, vb, O[c]);
        }
        const v16h eb = load_b_ntrans(&ldsE[buf][0], 32, lane);
        Oe = wmma_f16(pa, eb, Oe);

        buf ^= 1;
    }

    // ---- epilogue: normalize, store context (f16) + encoding (atomic f32)
#pragma unroll
    for (int r = 0; r < 8; ++r) {
        const float  inv  = 1.0f / lrun[r];
        const int    row  = q0 + r + 8 * half;
        const size_t orow = (size_t)b * L_ + row;
#pragma unroll
        for (int c = 0; c < 4; ++c)
            ctx16[orow * DIM_ + (size_t)h * DH_ + 16 * c + nl] = (_Float16)(O[c][r] * inv);
        atomicAdd(&encacc[orow * POS_ + nl], Oe[r] * inv * (1.0f / (float)H_));
    }
}

// ---------------------------------------------------------------------------
// Kernel 4: copy head-averaged encoding into output columns 1024..1039.
// ---------------------------------------------------------------------------
__global__ void enc_out_kernel(const float* __restrict__ encacc, float* __restrict__ out) {
    const int i = blockIdx.x * blockDim.x + threadIdx.x;   // [0, B*L*POS)
    const int p = i & 15;
    const int r = i >> 4;
    out[(size_t)r * OUTD_ + DIM_ + p] = encacc[i];
}

// ---------------------------------------------------------------------------
extern "C" void kernel_launch(void* const* d_in, const int* in_sizes, int n_in,
                              void* d_out, int out_size, void* d_ws, size_t ws_size,
                              hipStream_t stream) {
    (void)in_sizes; (void)n_in; (void)out_size; (void)ws_size;

    const float* q  = (const float*)d_in[0];
    const float* k  = (const float*)d_in[1];
    const float* v  = (const float*)d_in[2];
    const float* Wq = (const float*)d_in[3];
    const float* bq = (const float*)d_in[4];
    const float* Wk = (const float*)d_in[5];
    const float* bk = (const float*)d_in[6];
    const float* Wv = (const float*)d_in[7];
    const float* bv = (const float*)d_in[8];
    const float* Wo = (const float*)d_in[9];
    const float* bo = (const float*)d_in[10];
    float* out = (float*)d_out;

    // ---- carve workspace (256B aligned chunks) ----------------------------
    char* p = (char*)d_ws;
    auto carve = [&](size_t bytes) -> char* {
        char* r = p;
        p += (bytes + 255) & ~(size_t)255;
        return r;
    };
    const size_t BLD = (size_t)B_ * L_ * DIM_;      // 4 Mi elements
    const size_t WEL = (size_t)DIM_ * DIM_;         // 1 Mi elements

    _Float16* qf     = (_Float16*)carve(BLD * 2);
    _Float16* kf     = (_Float16*)carve(BLD * 2);
    _Float16* vf     = (_Float16*)carve(BLD * 2);
    _Float16* Wq16   = (_Float16*)carve(WEL * 2);
    _Float16* Wk16   = (_Float16*)carve(WEL * 2);
    _Float16* Wv16   = (_Float16*)carve(WEL * 2);
    _Float16* Wo16   = (_Float16*)carve(WEL * 2);
    _Float16* Qh     = (_Float16*)carve(BLD * 2);   // (b,h,l,d)
    _Float16* Kh     = (_Float16*)carve(BLD * 2);   // (b,h,l,d)
    _Float16* Vt     = (_Float16*)carve(BLD * 2);   // (b,h,d,l)
    _Float16* ctx16  = (_Float16*)carve(BLD * 2);   // (b*l, 1024)
    _Float16* encT   = (_Float16*)carve((size_t)B_ * POS_ * L_ * 2);
    float*    encacc = (float*)   carve((size_t)B_ * L_ * POS_ * 4);

    // 1) convert + relayout (covers B*L*DIM threads exactly)
    convert_kernel<<<(int)(BLD / 256), 256, 0, stream>>>(
        q, k, v, Wq, Wk, Wv, Wo, qf, kf, vf, Wq16, Wk16, Wv16, Wo16, encT, encacc);

    // 2) Q/K/V projections: 4096 waves of 16x64 tiles -> 1024 blocks x 4 waves
    proj_gemm<<<1024, 128, 0, stream>>>(qf, Wq16, bq, Qh, nullptr, 0);
    proj_gemm<<<1024, 128, 0, stream>>>(kf, Wk16, bk, Kh, nullptr, 1);
    proj_gemm<<<1024, 128, 0, stream>>>(vf, Wv16, bv, Vt, nullptr, 2);

    // 3) causal attention: B*H*(L/16) = 4096 waves
    attn_kernel<<<4096, 32, 0, stream>>>(Qh, Kh, Vt, encT, ctx16, encacc);

    // 4) output projection straight into the 1040-stride output buffer
    proj_gemm<<<1024, 128, 0, stream>>>(ctx16, Wo16, bo, nullptr, out, 3);

    // 5) encoding columns
    enc_out_kernel<<<(B_ * L_ * POS_) / 256, 256, 0, stream>>>(encacc, out);
}
